// AdditiveAttention_73993696575808
// MI455X (gfx1250) — matrix operand order repaired and checked by reading, not compiled, not run on hardware
//
#include <hip/hip_runtime.h>
#include <hip/hip_bf16.h>
#include <math.h>

// Problem constants: B=32, S=4096, D=512
#define BB 32
#define SS 4096
#define DD 512

typedef __attribute__((ext_vector_type(16))) __bf16        bf16x16;
typedef __attribute__((ext_vector_type(8)))  float         f32x8;
typedef __attribute__((ext_vector_type(4)))  float         f32x4;
typedef __attribute__((ext_vector_type(4)))  unsigned int  u32x4;

union TileAB { bf16x16 v; u32x4 q[2]; };

__device__ __forceinline__ unsigned short f2bf_rne(float f) {
    unsigned int u = __float_as_uint(f);
    unsigned int r = u + 0x7fffu + ((u >> 16) & 1u);
    return (unsigned short)(r >> 16);
}
__device__ __forceinline__ unsigned int pack_bf2(float lo, float hi) {
    return (unsigned int)f2bf_rne(lo) | ((unsigned int)f2bf_rne(hi) << 16);
}
// tanh(x) = 1 - 2/(exp(2x)+1); exp overflow -> inf -> +1, underflow -> 0 -> -1
__device__ __forceinline__ float fast_tanh(float x) {
    float e = __expf(2.0f * x);                        // v_exp_f32
    return 1.0f - 2.0f * __builtin_amdgcn_rcpf(e + 1.0f); // v_rcp_f32
}

// ---------------------------------------------------------------------------
// Kernel 1: W1k = W1[:, D:2D] -> bf16, stored N-MAJOR: w1kT[h*512 + k]
// WMMA B-operand friendly: for a 32x16 (KxN) bf16 B tile, lane L holds column
// N = L&15 and its K values are contiguous in memory.
// ---------------------------------------------------------------------------
__global__ void prep_w1kT(const float* __restrict__ W1,
                          unsigned short* __restrict__ w1kT) {
    int e = blockIdx.x * blockDim.x + threadIdx.x;   // 512*512 elements
    int k = e & 511;
    int h = e >> 9;
    w1kT[e] = f2bf_rne(W1[h * 1024 + 512 + k]);
}

// ---------------------------------------------------------------------------
// Kernel 2: qb1[b,h] = sum_q Q[0,b,q]*W1[h,q] + b1[h]   (32x512, tiny)
// ---------------------------------------------------------------------------
__global__ void prep_qb1(const float* __restrict__ Q,
                         const float* __restrict__ W1,
                         const float* __restrict__ b1,
                         float* __restrict__ qb1) {
    int e = blockIdx.x * blockDim.x + threadIdx.x;   // 32*512
    int b = e >> 9;
    int h = e & 511;
    const float* qrow = Q + b * 512;
    const float* wrow = W1 + h * 1024;
    float acc = b1[h];
    for (int q = 0; q < 512; ++q) acc += qrow[q] * wrow[q];
    qb1[e] = acc;
}

// ---------------------------------------------------------------------------
// Kernel 3 (main WMMA kernel):
//   score[b,s] = sum_h tanh( (K[b]@W1k^T)[s,h] + qb1[b,h] ) * W2[h] + b2
// One WG = (b, 64 s-rows). 8 waves; wave w owns h-columns [w*64, w*64+64)
// as 4 col-tiles and all 4 row-tiles -> 16 persistent 16x16 f32 accumulators.
// K slab (64 rows x 32 k) is DOUBLE-BUFFERED: next slab's global loads are
// issued after the B loads (in-order VMEM return => partial s_wait_loadcnt on
// B does not stall on them), converted+stored to the alternate LDS buffer at
// the loop tail. One barrier per k-step.
// ---------------------------------------------------------------------------
__global__ __launch_bounds__(256) void score_kernel(
    const float*          __restrict__ K,
    const unsigned short* __restrict__ w1kT,
    const float*          __restrict__ qb1,
    const float*          __restrict__ W2,
    const float*          __restrict__ b2,
    float*                __restrict__ score)
{
    __shared__ __align__(16) unsigned short Ks[2][64 * 32];  // 2 x 4 KB bf16
    __shared__ float scoreLDS[64];

    const int b    = blockIdx.x >> 6;          // 32 batches
    const int s0   = (blockIdx.x & 63) * 64;   // 64 s-blocks per batch
    const int tid  = threadIdx.x;
    const int lane = tid & 31;
    const int wv   = tid >> 5;
    const int half = lane >> 4;                // lane 0-15 vs 16-31
    const int l15  = lane & 15;

    if (tid < 64) scoreLDS[tid] = 0.0f;

    f32x8 acc[4][4];                            // [row-tile][col-tile]
    #pragma unroll
    for (int rt = 0; rt < 4; ++rt)
        #pragma unroll
        for (int ct = 0; ct < 4; ++ct)
            #pragma unroll
            for (int j = 0; j < 8; ++j)
                acc[rt][ct][j] = 0.0f;

    // bf16 A-matrix (16x32) per-lane K offsets: lanes 0-15 -> {0..7,16..23},
    // lanes 16-31 -> {8..15,24..31}
    const int offA0 = half ? 8 : 0;
    const int offA1 = half ? 24 : 16;

    // staging assignment: thread -> (row, 8-col chunk)
    const int r  = tid >> 2;
    const int c0 = (tid & 3) * 8;
    const float* Krow = K + ((size_t)(b * SS + s0 + r)) * DD + c0;

    // ---- prologue: load + convert + store slab 0 into buffer 0 ----
    {
        f32x4 a = *(const f32x4*)(Krow);
        f32x4 c = *(const f32x4*)(Krow + 4);
        u32x4 p;
        p.x = pack_bf2(a.x, a.y);
        p.y = pack_bf2(a.z, a.w);
        p.z = pack_bf2(c.x, c.y);
        p.w = pack_bf2(c.z, c.w);
        *(u32x4*)(&Ks[0][r * 32 + c0]) = p;
    }

    for (int i = 0; i < 16; ++i) {
        __syncthreads();                       // buf[i&1] ready for all waves
        const int k0 = i * 32;

        // ---- B tiles for this k-step (L2-resident bf16 W1k) ----
        TileAB Bt[4];
        const size_t krow = (size_t)k0;   // lane&15 = N column of B
        #pragma unroll
        for (int ct = 0; ct < 4; ++ct) {
            const unsigned short* p = w1kT + (size_t)(wv * 64 + ct * 16 + l15) * DD + krow;
            Bt[ct].q[0] = *(const u32x4*)(p + offA0);
            Bt[ct].q[1] = *(const u32x4*)(p + offA1);
        }

        // ---- issue next K slab's global loads (after B: in-order VMEM
        //      return means WMMA's partial wait on B won't stall on these) ----
        f32x4 ra, rb;
        if (i < 15) {
            ra = *(const f32x4*)(Krow + k0 + 32);
            rb = *(const f32x4*)(Krow + k0 + 36);
            __builtin_prefetch(w1kT + krow + 32 * DD + wv * 64, 0, 1);
        }

        // ---- 16 WMMAs: 4 row-tiles x 4 col-tiles ----
        const unsigned short* buf = &Ks[i & 1][0];
        #pragma unroll
        for (int rt = 0; rt < 4; ++rt) {
            TileAB At;
            const unsigned short* ap = buf + (rt * 16 + l15) * 32;
            At.q[0] = *(const u32x4*)(ap + offA0);
            At.q[1] = *(const u32x4*)(ap + offA1);
            #pragma unroll
            for (int ct = 0; ct < 4; ++ct) {
                acc[rt][ct] = __builtin_amdgcn_wmma_f32_16x16x32_bf16(
                    false, At.v, false, Bt[ct].v,
                    (short)0, acc[rt][ct], false, false);
            }
        }

        // ---- convert + store next slab into the other buffer ----
        if (i < 15) {
            u32x4 p;
            p.x = pack_bf2(ra.x, ra.y);
            p.y = pack_bf2(ra.z, ra.w);
            p.z = pack_bf2(rb.x, rb.y);
            p.w = pack_bf2(rb.z, rb.w);
            *(u32x4*)(&Ks[(i + 1) & 1][r * 32 + c0]) = p;
        }
    }

    // ---- epilogue: tanh, *W2, reduce over h ----
    // C layout: lane L -> N = L&15 ; VGPR j -> M = j + (L>=16 ? 8 : 0)
    float prt[4][8];
    #pragma unroll
    for (int rt = 0; rt < 4; ++rt)
        #pragma unroll
        for (int j = 0; j < 8; ++j) prt[rt][j] = 0.0f;

    #pragma unroll
    for (int ct = 0; ct < 4; ++ct) {
        const int n = wv * 64 + ct * 16 + l15;     // global h column
        const float qv  = qb1[b * DD + n];
        const float w2v = W2[n];
        #pragma unroll
        for (int rt = 0; rt < 4; ++rt)
            #pragma unroll
            for (int j = 0; j < 8; ++j)
                prt[rt][j] += fast_tanh(acc[rt][ct][j] + qv) * w2v;
    }
    // sum across the 16 lanes of each half (same M rows within a half)
    #pragma unroll
    for (int m = 1; m < 16; m <<= 1)
        #pragma unroll
        for (int rt = 0; rt < 4; ++rt)
            #pragma unroll
            for (int j = 0; j < 8; ++j)
                prt[rt][j] += __shfl_xor(prt[rt][j], m, 32);

    if (l15 == 0) {
        #pragma unroll
        for (int rt = 0; rt < 4; ++rt)
            #pragma unroll
            for (int j = 0; j < 8; ++j)
                atomicAdd(&scoreLDS[rt * 16 + half * 8 + j], prt[rt][j]);
    }
    __syncthreads();
    if (tid < 64) score[b * SS + s0 + tid] = scoreLDS[tid] + b2[0];
}

// ---------------------------------------------------------------------------
// Kernel 4: per-batch softmax over S, emit attn_weight, then attn @ V
// ---------------------------------------------------------------------------
__global__ __launch_bounds__(256) void softmax_av_kernel(
    const float* __restrict__ score,
    const float* __restrict__ V,
    float*       __restrict__ attnA,
    float*       __restrict__ out_attn)
{
    __shared__ float w[SS];          // 16 KB
    __shared__ float red[256];
    const int b = blockIdx.x;
    const int t = threadIdx.x;

    float m = -3.4e38f;
    for (int i = t; i < SS; i += 256) { float v = score[b * SS + i]; w[i] = v; m = fmaxf(m, v); }
    red[t] = m; __syncthreads();
    for (int o = 128; o > 0; o >>= 1) { if (t < o) red[t] = fmaxf(red[t], red[t + o]); __syncthreads(); }
    float mx = red[0];
    __syncthreads();

    float s = 0.0f;
    for (int i = t; i < SS; i += 256) { float e = __expf(w[i] - mx); w[i] = e; s += e; }
    red[t] = s; __syncthreads();
    for (int o = 128; o > 0; o >>= 1) { if (t < o) red[t] += red[t + o]; __syncthreads(); }
    float inv = 1.0f / red[0];
    __syncthreads();

    for (int i = t; i < SS; i += 256) {
        float wa = w[i] * inv;
        w[i] = wa;
        out_attn[b * SS + i] = wa;
    }
    __syncthreads();

    // attn_applied[b, d] = sum_s w[s] * V[b, s, d]; thread owns d = t, t+256
    float a0 = 0.0f, a1 = 0.0f;
    const float* Vb = V + (size_t)b * SS * DD;
    #pragma unroll 4
    for (int si = 0; si < SS; ++si) {
        float wv = w[si];
        a0 += wv * Vb[(size_t)si * DD + t];
        a1 += wv * Vb[(size_t)si * DD + t + 256];
    }
    attnA[b * DD + t]       = a0;
    attnA[b * DD + t + 256] = a1;
}

// ---------------------------------------------------------------------------
// Kernel 5: out[b,h] = x[b]@Wc[:, :D]^T + attnA[b]@Wc[:, D:]^T + bc
// ---------------------------------------------------------------------------
__global__ void out_kernel(const float* __restrict__ x,
                           const float* __restrict__ attnA,
                           const float* __restrict__ Wc,
                           const float* __restrict__ bc,
                           float* __restrict__ out)
{
    int e = blockIdx.x * blockDim.x + threadIdx.x;   // 32*512
    int b = e >> 9;
    int h = e & 511;
    const float* xr = x + b * DD;
    const float* ar = attnA + b * DD;
    const float* w0 = Wc + h * 1024;
    const float* w1 = w0 + 512;
    float acc = bc[h];
    for (int d = 0; d < DD; ++d) acc += xr[d] * w0[d] + ar[d] * w1[d];
    out[e] = acc;
}

// ---------------------------------------------------------------------------
extern "C" void kernel_launch(void* const* d_in, const int* in_sizes, int n_in,
                              void* d_out, int out_size, void* d_ws, size_t ws_size,
                              hipStream_t stream) {
    (void)in_sizes; (void)n_in; (void)out_size; (void)ws_size;
    const float* x  = (const float*)d_in[0];
    const float* Q  = (const float*)d_in[1];
    const float* K  = (const float*)d_in[2];
    const float* V  = (const float*)d_in[3];
    const float* W1 = (const float*)d_in[4];
    const float* b1 = (const float*)d_in[5];
    const float* W2 = (const float*)d_in[6];
    const float* b2 = (const float*)d_in[7];
    const float* Wc = (const float*)d_in[8];
    const float* bc = (const float*)d_in[9];

    float* out      = (float*)d_out;            // (B,1,D) = 16384 floats
    float* out_attn = out + BB * DD;            // (B,1,S) = 131072 floats

    // workspace layout
    char* ws = (char*)d_ws;
    unsigned short* w1kT = (unsigned short*)ws;                       // 512 KB
    float* qb1   = (float*)(ws + 524288);                             //  64 KB
    float* scr   = (float*)(ws + 524288 + 65536);                     // 512 KB
    float* attnA = (float*)(ws + 524288 + 65536 + 524288);            //  64 KB

    prep_w1kT<<<1024, 256, 0, stream>>>(W1, w1kT);
    prep_qb1 <<<  64, 256, 0, stream>>>(Q, W1, b1, qb1);
    score_kernel<<<BB * (SS / 64), 256, 0, stream>>>(K, w1kT, qb1, W2, b2, scr);
    softmax_av_kernel<<<BB, 256, 0, stream>>>(scr, V, attnA, out_attn);
    out_kernel<<<64, 256, 0, stream>>>(x, attnA, Wc, bc, out);
}
